// MoleculeModel_15444702396778
// MI455X (gfx1250) — compile-verified, matrix operand order checked by reading, not run on hardware
//
#include <hip/hip_runtime.h>
#include <hip/hip_bf16.h>

// ---------------------------------------------------------------------------
// MoleculeModel head for MI455X (gfx1250, wave32, WMMA)
//
// Phase 1 (memory-bound, 512MB stream):  fused sigmoid-weight + weighted
//   segment-sum pooling. One wave per 64-atom chunk (segment_ids sorted),
//   per-task accumulators in registers, atomic f32 flush at segment edges.
// Phase 2 (matrix-core-bound, 78.6 GFLOP): per-(128-graph, task) tile MLP,
//   all activations resident in LDS as f16, v_wmma_f32_16x16x32_f16.
// ---------------------------------------------------------------------------

typedef __attribute__((ext_vector_type(16))) _Float16 v16h;
typedef __attribute__((ext_vector_type(8)))  float    v8f;

#define T_TASKS 12
#define FDIM    128
#define HDIM    128
#define XP      136   // LDS pitch (halfs) for activation tiles
#define WP      136   // LDS pitch (halfs) for transposed weights
#define BN_EPS  1e-5f

// ---------------------------------------------------------------------------
// Kernel 0: zero the mol accumulator in workspace
// ---------------------------------------------------------------------------
__global__ void zero_f4_kernel(float4* __restrict__ p, long n4) {
    long i = (long)blockIdx.x * blockDim.x + threadIdx.x;
    long stride = (long)gridDim.x * blockDim.x;
    float4 z = {0.f, 0.f, 0.f, 0.f};
    for (; i < n4; i += stride) p[i] = z;
}

// ---------------------------------------------------------------------------
// Kernel 1: fused WeightAndSum pooling.
//   w[n,t] = sigmoid(x[n]·atom_w[t] + atom_b[t])
//   mol[g,t,f] += w[n,t] * x[n,f]
// One wave per contiguous chunk of 64 atoms. Lane l owns features [4l,4l+4).
// ---------------------------------------------------------------------------
__global__ void __launch_bounds__(256)
pool_kernel(const float* __restrict__ nf, const int* __restrict__ seg,
            const float* __restrict__ atom_w, const float* __restrict__ atom_b,
            float* __restrict__ mol, int N) {
    const int tid  = blockIdx.x * blockDim.x + threadIdx.x;
    const int wave = tid >> 5;
    const int lane = tid & 31;
    const long base = (long)wave * 64;
    if (base >= (long)N) return;
    const long end = (base + 64 < (long)N) ? base + 64 : (long)N;

    // per-lane slice of the task weight matrix: aw[t] = atom_w[t, 4l..4l+3]
    float4 aw[T_TASKS];
    float  ab[T_TASKS];
#pragma unroll
    for (int t = 0; t < T_TASKS; ++t) {
        aw[t] = reinterpret_cast<const float4*>(atom_w + t * FDIM)[lane];
        ab[t] = atom_b[t];
    }

    float acc[T_TASKS][4];
#pragma unroll
    for (int t = 0; t < T_TASKS; ++t)
#pragma unroll
        for (int j = 0; j < 4; ++j) acc[t][j] = 0.f;

    int segCur = -1;

    auto flush = [&]() {
#pragma unroll
        for (int t = 0; t < T_TASKS; ++t) {
            float* p = mol + ((size_t)segCur * T_TASKS + t) * FDIM + lane * 4;
            atomicAdd(p + 0, acc[t][0]);
            atomicAdd(p + 1, acc[t][1]);
            atomicAdd(p + 2, acc[t][2]);
            atomicAdd(p + 3, acc[t][3]);
        }
    };

    for (long n = base; n < end; ++n) {
        int s = seg[n];                        // uniform across the wave
        if (s != segCur) {
            if (segCur >= 0) flush();
            segCur = s;
#pragma unroll
            for (int t = 0; t < T_TASKS; ++t)
#pragma unroll
                for (int j = 0; j < 4; ++j) acc[t][j] = 0.f;
        }
        // stream the row; prefetch a few rows ahead (global_prefetch_b8)
        __builtin_prefetch(nf + (n + 4) * FDIM + lane * 4, 0, 0);
        float4 x = reinterpret_cast<const float4*>(nf + n * FDIM)[lane];

        // 12 per-lane partial dots, then wave32 butterfly reduce
        float d[T_TASKS];
#pragma unroll
        for (int t = 0; t < T_TASKS; ++t)
            d[t] = x.x * aw[t].x + x.y * aw[t].y + x.z * aw[t].z + x.w * aw[t].w;
#pragma unroll
        for (int off = 16; off > 0; off >>= 1)
#pragma unroll
            for (int t = 0; t < T_TASKS; ++t)
                d[t] += __shfl_xor(d[t], off, 32);

#pragma unroll
        for (int t = 0; t < T_TASKS; ++t) {
            float w = 1.f / (1.f + __expf(-(d[t] + ab[t])));  // sigmoid
            acc[t][0] += w * x.x;
            acc[t][1] += w * x.y;
            acc[t][2] += w * x.z;
            acc[t][3] += w * x.w;
        }
    }
    if (segCur >= 0) flush();
}

// ---------------------------------------------------------------------------
// WMMA fragment loaders per CDNA5 ISA layouts (wave32).
// A 16x32 f16: lane 0-15 -> M=lane, K in {0..7,16..23}; lane 16-31 same M,
//              K in {8..15,24..31}; 2 halfs per VGPR.
// B 32x16 f16: lane = N column, K pairs per VGPR (same split) -> load from
//              an N-major (transposed) weight tile so pairs are contiguous.
// C/D 16x16 f32: lane 0-15 -> N=lane, M=v; lane 16-31 -> N=lane-16, M=v+8.
// ---------------------------------------------------------------------------
__device__ __forceinline__ v16h load_a_frag(const _Float16* __restrict__ X,
                                            int mBase, int kBase, int lane) {
    v16h a;
    const int hs = (lane >> 4) & 1;
    const _Float16* row = X + (mBase + (lane & 15)) * XP;
#pragma unroll
    for (int v = 0; v < 8; ++v) {
        int k = kBase + (v >> 2) * 16 + hs * 8 + (v & 3) * 2;
        a[2 * v]     = row[k];
        a[2 * v + 1] = row[k + 1];
    }
    return a;
}

__device__ __forceinline__ v16h load_b_frag(const _Float16* __restrict__ Wt,
                                            int kBase, int nBase, int lane) {
    v16h b;
    const int hs = (lane >> 4) & 1;
    const _Float16* row = Wt + (nBase + (lane & 15)) * WP;  // N-major
#pragma unroll
    for (int v = 0; v < 8; ++v) {
        int k = kBase + (v >> 2) * 16 + hs * 8 + (v & 3) * 2;
        b[2 * v]     = row[k];
        b[2 * v + 1] = row[k + 1];
    }
    return b;
}

// ---------------------------------------------------------------------------
// One fused layer: X_out = relu(X_in @ W + b) [ * s + c  if BN ]
// 128x128 tile, 8 waves, each wave: 16 rows x 128 cols = 8 Ntiles x 4 Ksteps
// = 32 v_wmma_f32_16x16x32_f16.
// ---------------------------------------------------------------------------
template <bool BN>
__device__ __forceinline__ void fc_layer(
    const _Float16* __restrict__ sXin, _Float16* __restrict__ sXout,
    _Float16* __restrict__ sWt, float* sB, float* sS, float* sC,
    const float* __restrict__ Wg, const float* __restrict__ bg,
    const float* __restrict__ gg, const float* __restrict__ beg,
    const float* __restrict__ rmg, const float* __restrict__ rvg, int tid) {
    // stage weights (f32 global, row-major [in][out]) -> f16 LDS, N-major
    for (int i = tid; i < HDIM * HDIM; i += 256) {
        int r = i >> 7;       // input dim (K)
        int c = i & 127;      // output dim (N)
        sWt[c * WP + r] = (_Float16)Wg[i];
    }
    if (tid < HDIM) {
        sB[tid] = bg[tid];
        if (BN) {
            float s = gg[tid] * rsqrtf(rvg[tid] + BN_EPS);
            sS[tid] = s;
            sC[tid] = beg[tid] - rmg[tid] * s;
        }
    }
    __syncthreads();

    const int wid = tid >> 5, lane = tid & 31;
    const int mBase = wid << 4;

    v8f acc[8];
#pragma unroll
    for (int n = 0; n < 8; ++n) acc[n] = (v8f){};

#pragma unroll
    for (int k = 0; k < 4; ++k) {
        v16h a = load_a_frag(sXin, mBase, k * 32, lane);
#pragma unroll
        for (int n = 0; n < 8; ++n) {
            v16h b = load_b_frag(sWt, k * 32, n * 16, lane);
            acc[n] = __builtin_amdgcn_wmma_f32_16x16x32_f16(
                false, a, false, b, (short)0, acc[n], false, false);
        }
    }

    // epilogue: bias + relu (+ folded BN), store f16
    const int nl = lane & 15;
    const int mOff = (lane >> 4) << 3;
#pragma unroll
    for (int n = 0; n < 8; ++n) {
        int col = (n << 4) + nl;
        float bias = sB[col];
        float sc = BN ? sS[col] : 1.0f;
        float sh = BN ? sC[col] : 0.0f;
#pragma unroll
        for (int v = 0; v < 8; ++v) {
            float y = acc[n][v] + bias;
            y = fmaxf(y, 0.f);
            if (BN) y = y * sc + sh;
            sXout[(mBase + mOff + v) * XP + col] = (_Float16)y;
        }
    }
    __syncthreads();
}

// ---------------------------------------------------------------------------
// Kernel 2: per-(graph-tile, task) MLP head. Grid = (ceil(G/128), T).
// ---------------------------------------------------------------------------
__global__ void __launch_bounds__(256)
mlp_kernel(const float* __restrict__ mol,
           const float* __restrict__ W1, const float* __restrict__ b1,
           const float* __restrict__ g1, const float* __restrict__ be1,
           const float* __restrict__ rm1, const float* __restrict__ rv1,
           const float* __restrict__ W2, const float* __restrict__ b2,
           const float* __restrict__ g2, const float* __restrict__ be2,
           const float* __restrict__ rm2, const float* __restrict__ rv2,
           const float* __restrict__ W3, const float* __restrict__ b3,
           const float* __restrict__ g3, const float* __restrict__ be3,
           const float* __restrict__ rm3, const float* __restrict__ rv3,
           const float* __restrict__ oW1, const float* __restrict__ ob1,
           const float* __restrict__ oW2, const float* __restrict__ ob2,
           float* __restrict__ pred, int G) {
    __shared__ _Float16 sXa[128 * XP];
    __shared__ _Float16 sXb[128 * XP];
    __shared__ _Float16 sWt[128 * WP];
    __shared__ float sB[HDIM], sS[HDIM], sC[HDIM];

    const int tid = threadIdx.x;
    const int gBase = blockIdx.x * 128;
    const int t = blockIdx.y;

    const size_t wOff = (size_t)t * HDIM * HDIM;
    const size_t vOff = (size_t)t * HDIM;

    // load mol tile [128 graphs x 128 feats] for this task, f32 -> f16
    for (int i = tid; i < 128 * FDIM; i += 256) {
        int r = i >> 7, f = i & 127;
        int g = gBase + r;
        float v = (g < G) ? mol[((size_t)g * T_TASKS + t) * FDIM + f] : 0.f;
        sXa[r * XP + f] = (_Float16)v;
    }

    fc_layer<true>(sXa, sXb, sWt, sB, sS, sC, W1 + wOff, b1 + vOff,
                   g1 + vOff, be1 + vOff, rm1 + vOff, rv1 + vOff, tid);
    fc_layer<true>(sXb, sXa, sWt, sB, sS, sC, W2 + wOff, b2 + vOff,
                   g2 + vOff, be2 + vOff, rm2 + vOff, rv2 + vOff, tid);
    fc_layer<true>(sXa, sXb, sWt, sB, sS, sC, W3 + wOff, b3 + vOff,
                   g3 + vOff, be3 + vOff, rm3 + vOff, rv3 + vOff, tid);
    fc_layer<false>(sXb, sXa, sWt, sB, sS, sC, oW1 + wOff, ob1 + vOff,
                    nullptr, nullptr, nullptr, nullptr, tid);

    // final projection: pred[g,t] = z . oW2[t] + ob2[t]   (Z f16 in sXa)
    if (tid < HDIM) sS[tid] = oW2[vOff + tid];
    __syncthreads();
    if (tid < 128) {
        int g = gBase + tid;
        if (g < G) {
            const _Float16* row = sXa + tid * XP;
            float s = 0.f;
#pragma unroll 4
            for (int h = 0; h < HDIM; ++h) s += (float)row[h] * sS[h];
            pred[(size_t)g * T_TASKS + t] = s + ob2[t];
        }
    }
}

// ---------------------------------------------------------------------------
// Host launcher
// ---------------------------------------------------------------------------
extern "C" void kernel_launch(void* const* d_in, const int* in_sizes, int n_in,
                              void* d_out, int out_size, void* d_ws, size_t ws_size,
                              hipStream_t stream) {
    const float* node_feats = (const float*)d_in[0];
    const int*   seg        = (const int*)d_in[1];
    // d_in[2] = num_graphs (derived on host instead)
    const float* atom_w = (const float*)d_in[3];
    const float* atom_b = (const float*)d_in[4];
    const float *W1 = (const float*)d_in[5],  *b1 = (const float*)d_in[6];
    const float *g1 = (const float*)d_in[7],  *be1 = (const float*)d_in[8];
    const float *rm1 = (const float*)d_in[9], *rv1 = (const float*)d_in[10];
    const float *W2 = (const float*)d_in[11], *b2 = (const float*)d_in[12];
    const float *g2 = (const float*)d_in[13], *be2 = (const float*)d_in[14];
    const float *rm2 = (const float*)d_in[15], *rv2 = (const float*)d_in[16];
    const float *W3 = (const float*)d_in[17], *b3 = (const float*)d_in[18];
    const float *g3 = (const float*)d_in[19], *be3 = (const float*)d_in[20];
    const float *rm3 = (const float*)d_in[21], *rv3 = (const float*)d_in[22];
    const float *oW1 = (const float*)d_in[23], *ob1 = (const float*)d_in[24];
    const float *oW2 = (const float*)d_in[25], *ob2 = (const float*)d_in[26];

    const int N = in_sizes[0] / FDIM;       // 1,000,000
    const int G = out_size / T_TASKS;       // 50,000

    float* mol  = (float*)d_ws;             // [G, T, F] f32 accumulator
    float* pred = (float*)d_out;            // [G, T]

    // 0) zero mol
    long n4 = (long)G * T_TASKS * FDIM / 4;
    zero_f4_kernel<<<2048, 256, 0, stream>>>((float4*)mol, n4);

    // 1) fused pooling: one wave per 64-atom chunk, 8 waves per block
    int waves   = (N + 63) / 64;
    int blocks1 = (waves + 7) / 8;
    pool_kernel<<<blocks1, 256, 0, stream>>>(node_feats, seg, atom_w, atom_b,
                                             mol, N);

    // 2) WMMA MLP head: (graph-tile, task) grid
    dim3 grid2((G + 127) / 128, T_TASKS);
    mlp_kernel<<<grid2, 256, 0, stream>>>(mol,
                                          W1, b1, g1, be1, rm1, rv1,
                                          W2, b2, g2, be2, rm2, rv2,
                                          W3, b3, g3, be3, rm3, rv3,
                                          oW1, ob1, oW2, ob2,
                                          pred, G);
}
